// TransformerEncoder_59717225284218
// MI455X (gfx1250) — matrix operand order repaired and checked by reading, not compiled, not run on hardware
//
#include <hip/hip_runtime.h>
#include <math.h>

// ---------------------------------------------------------------------------
// TransformerEncoder (dead-code-eliminated) for MI455X / gfx1250.
//   qraw = x @ Wq + bq ; kraw = source @ Wk + bk
//   s    = l2norm(qraw) + l2norm(kraw)           (idx gather is identity)
//   oraw = s @ Wf + bf ; h = l2norm(oraw + x)
//   t    = gelu(h @ W1 + b1) ; out = t @ W2 + b2 + h
// All GEMMs: f16 operands (A activations, W pre-transposed to [N][K]),
// f32 WMMA accumulate, TDM double-buffered LDS tile fills.
// ---------------------------------------------------------------------------

typedef __attribute__((ext_vector_type(16))) _Float16 v16h;
typedef __attribute__((ext_vector_type(8)))  _Float16 v8h;
typedef __attribute__((ext_vector_type(4)))  _Float16 v4h;
typedef __attribute__((ext_vector_type(8)))  float    v8f;
typedef __attribute__((ext_vector_type(4)))  unsigned int v4u;
typedef __attribute__((ext_vector_type(4)))  int      v4i;
typedef __attribute__((ext_vector_type(8)))  int      v8i;

#define BM 64
#define BN 128
#define BK 32
#define ASTR 40   // halves: 32 data + 8 pad (80B rows keep 16B alignment)
#define BSTR 40

#define EPI_NONE 0
#define EPI_GELU 1
#define EPI_ADD  2

#if __has_builtin(__builtin_amdgcn_tensor_load_to_lds) && __has_builtin(__builtin_amdgcn_s_wait_tensorcnt)
#define USE_TDM 1
#else
#define USE_TDM 0
#endif

__device__ __forceinline__ float gelu_exact(float v) {
    return 0.5f * v * (1.0f + erff(v * 0.70710678118654752f));
}

#if USE_TDM
// Issue one TDM 2D tile load: rows x BK(32) f16 elements, row stride
// row_stride_elts in global, packed into LDS with 4-DWORD pad every
// 16 DWORDs (-> ASTR/BSTR = 40-half rows).  D# per CDNA5 ISA ch.8.
__device__ __forceinline__ void tdm_load_tile(const _Float16* gsrc, unsigned lds_byte,
                                              int rows, int row_stride_elts) {
    unsigned long long ga = (unsigned long long)(const void*)gsrc;
    v4u g0;
    g0.x = 1u;                                            // count=1, user mode
    g0.y = lds_byte;                                      // lds_addr
    g0.z = (unsigned)ga;                                  // global_addr[31:0]
    g0.w = (unsigned)((ga >> 32) & 0x01FFFFFFu) | (2u << 30); // addr[56:32] | type=2
    v8i g1;
    g1[0] = (1 << 16)        // data_size = 2 bytes
          | (1 << 20)        // pad_enable
          | (3 << 22)        // pad_interval: 16 DWORDs (one 32-half row)
          | (3 << 25);       // pad_amount:   4 DWORDs (8 halves)
    g1[1] = (row_stride_elts & 0xFFFF) << 16;                               // tensor_dim0 lo
    g1[2] = ((row_stride_elts >> 16) & 0xFFFF) | ((rows & 0xFFFF) << 16);   // dim0 hi | dim1 lo
    g1[3] = (((unsigned)rows >> 16) & 0xFFFF) | (BK << 16);                 // dim1 hi | tile_dim0
    g1[4] = rows & 0xFFFF;                                                  // tile_dim1 (tile_dim2=0)
    g1[5] = row_stride_elts;                                                // tensor_dim0_stride lo
    g1[6] = 0;
    g1[7] = 0;
    v4i z4 = {0, 0, 0, 0};
#if __clang_major__ >= 23
    v8i z8 = {0, 0, 0, 0, 0, 0, 0, 0};
    __builtin_amdgcn_tensor_load_to_lds(g0, g1, z4, z4, z8, 0);
#else
    __builtin_amdgcn_tensor_load_to_lds(g0, g1, z4, z4, 0);
#endif
}
#endif

// C[M,N] = op( A[M,K](f16) @ Wt[N,K]^T(f16) + bias[N] )  (+ extra for EPI_ADD)
template <int EPI, typename OUT>
__global__ __launch_bounds__(256)
void gemm_wmma(const _Float16* __restrict__ A, const _Float16* __restrict__ Wt,
               const float* __restrict__ bias, OUT* __restrict__ C,
               int M, int N, int K, const _Float16* __restrict__ extra) {
    __shared__ __align__(16) _Float16 As[2][BM * ASTR];   // 2 x 5 KB
    __shared__ __align__(16) _Float16 Bs[2][BN * BSTR];   // 2 x 10 KB

    const int tid  = threadIdx.x;
    const int lane = tid & 31;
    const int wave = tid >> 5;
    const int mW   = (wave >> 2) * 32;
    const int nW   = (wave & 3) * 32;
    const int mBase = blockIdx.y * BM;
    const int nBase = blockIdx.x * BN;
    const int l15 = lane & 15;
    const int lh  = lane >> 4;
    const int kbA = lh * 8;            // A frag K base (halves 0..7 / 8..15 split)
    const int kbB = lh * 8;            // B frag K base (same split as A)

    const _Float16* Ablk = A  + (size_t)mBase * (size_t)K;
    const _Float16* Bblk = Wt + (size_t)nBase * (size_t)K;

    v8f acc[2][2] = {};
    const int nIter = K / BK;

#if USE_TDM
    const unsigned ldsA0 = (unsigned)(unsigned long long)(const void*)&As[0][0];
    const unsigned ldsA1 = (unsigned)(unsigned long long)(const void*)&As[1][0];
    const unsigned ldsB0 = (unsigned)(unsigned long long)(const void*)&Bs[0][0];
    const unsigned ldsB1 = (unsigned)(unsigned long long)(const void*)&Bs[1][0];
    if (wave == 0)      tdm_load_tile(Ablk, ldsA0, BM, K);
    else if (wave == 1) tdm_load_tile(Bblk, ldsB0, BN, K);
#endif

    for (int it = 0; it < nIter; ++it) {
        const int cur = it & 1;
#if USE_TDM
        if (it + 1 < nIter) {   // prefetch next tiles into the other buffer
            const int k1 = (it + 1) * BK;
            if (wave == 0)      tdm_load_tile(Ablk + k1, cur ? ldsA0 : ldsA1, BM, K);
            else if (wave == 1) tdm_load_tile(Bblk + k1, cur ? ldsB0 : ldsB1, BN, K);
        }
        if (it + 1 < nIter) __builtin_amdgcn_s_wait_tensorcnt(1);
        else                __builtin_amdgcn_s_wait_tensorcnt(0);
#else
        {   // synchronous cooperative fill (fallback): pure b128 copies
            const int k0 = it * BK;
            { const int r = tid >> 2, s = tid & 3;
              *(v8h*)&As[cur][r * ASTR + s * 8] =
                  *(const v8h*)&Ablk[(size_t)r * K + k0 + s * 8]; }
            #pragma unroll
            for (int hseg = 0; hseg < 2; ++hseg) {
                const int seg = tid + hseg * 256;
                const int r = seg >> 2, s = seg & 3;
                *(v8h*)&Bs[cur][r * BSTR + s * 8] =
                    *(const v8h*)&Bblk[(size_t)r * K + k0 + s * 8];
            }
        }
#endif
        __syncthreads();   // tiles visible to all waves

        // Fragments per ISA VGPR layouts:
        //  A 16x32: lane l -> row l&15; halves {K=(l>>4)*8+0..7, K=16+(l>>4)*8+0..7}
        //  B 32x16: lane l -> col l&15; halves {K=(l>>4)*8+0..7, K=16+(l>>4)*8+0..7}
        union FragU { v16h v; v8h h[2]; };
        v16h afrag[2], bfrag[2];
        #pragma unroll
        for (int ms = 0; ms < 2; ++ms) {
            const int row = mW + ms * 16 + l15;
            FragU u;
            u.h[0] = *(const v8h*)&As[cur][row * ASTR + kbA];
            u.h[1] = *(const v8h*)&As[cur][row * ASTR + 16 + kbA];
            afrag[ms] = u.v;
        }
        #pragma unroll
        for (int ns = 0; ns < 2; ++ns) {
            const int col = nW + ns * 16 + l15;
            FragU u;
            u.h[0] = *(const v8h*)&Bs[cur][col * BSTR + kbB];
            u.h[1] = *(const v8h*)&Bs[cur][col * BSTR + 16 + kbB];
            bfrag[ns] = u.v;
        }
        #pragma unroll
        for (int ms = 0; ms < 2; ++ms)
            #pragma unroll
            for (int ns = 0; ns < 2; ++ns)
                acc[ms][ns] = __builtin_amdgcn_wmma_f32_16x16x32_f16(
                    false, afrag[ms], false, bfrag[ns],
                    (short)0, acc[ms][ns], false, false);

        __syncthreads();   // all reads of buf `cur` done before it is refilled
    }

    // Epilogue: C/D layout lane l -> n = l&15, VGPR j -> m = (l>>4)*8 + j
    #pragma unroll
    for (int ms = 0; ms < 2; ++ms) {
        #pragma unroll
        for (int ns = 0; ns < 2; ++ns) {
            const int col = nBase + nW + ns * 16 + l15;
            const float bv = bias[col];
            #pragma unroll
            for (int j = 0; j < 8; ++j) {
                const int row = mBase + mW + ms * 16 + lh * 8 + j;
                float v = acc[ms][ns][j] + bv;
                if (EPI == EPI_GELU) v = gelu_exact(v);
                if (EPI == EPI_ADD)  v += (float)extra[(size_t)row * N + col];
                C[(size_t)row * N + col] = (OUT)v;
            }
        }
    }
}

// f32 -> f16 elementwise convert (n multiple of 1024; one float4 per thread)
__global__ __launch_bounds__(256)
void cvt_f2h(const float* __restrict__ in, _Float16* __restrict__ out) {
    const size_t i = (size_t)blockIdx.x * 256 + threadIdx.x;
    float4 f = ((const float4*)in)[i];
    v4h o = { (_Float16)f.x, (_Float16)f.y, (_Float16)f.z, (_Float16)f.w };
    ((v4h*)out)[i] = o;
}

// Wt[n*K + k] = (f16) W[k*N + n]   (grid covers K*N exactly)
__global__ __launch_bounds__(256)
void transpose_f2h(const float* __restrict__ W, _Float16* __restrict__ Wt, int K, int N) {
    const int idx = blockIdx.x * 256 + threadIdx.x;
    const int k = idx % K, n = idx / K;
    Wt[idx] = (_Float16)W[(size_t)k * N + n];
}

// s = l2norm(q_row) + l2norm(k_row), D=512, one 256-thread block per row
__global__ __launch_bounds__(256)
void l2norm_sum2(const _Float16* __restrict__ q, const _Float16* __restrict__ k,
                 _Float16* __restrict__ s) {
    __shared__ float red[256];
    const int row = blockIdx.x, tid = threadIdx.x;
    const _Float16* qr = q + (size_t)row * 512;
    const _Float16* kr = k + (size_t)row * 512;
    float q0 = (float)qr[tid], q1 = (float)qr[tid + 256];
    float k0 = (float)kr[tid], k1 = (float)kr[tid + 256];

    red[tid] = q0 * q0 + q1 * q1;
    __syncthreads();
    for (int st = 128; st > 0; st >>= 1) { if (tid < st) red[tid] += red[tid + st]; __syncthreads(); }
    const float invq = 1.0f / fmaxf(sqrtf(red[0]), 1e-12f);
    __syncthreads();
    red[tid] = k0 * k0 + k1 * k1;
    __syncthreads();
    for (int st = 128; st > 0; st >>= 1) { if (tid < st) red[tid] += red[tid + st]; __syncthreads(); }
    const float invk = 1.0f / fmaxf(sqrtf(red[0]), 1e-12f);

    s[(size_t)row * 512 + tid]       = (_Float16)(q0 * invq + k0 * invk);
    s[(size_t)row * 512 + tid + 256] = (_Float16)(q1 * invq + k1 * invk);
}

// h = l2norm(oraw_row + x_row), D=512
__global__ __launch_bounds__(256)
void add_l2norm(const _Float16* __restrict__ oraw, const float* __restrict__ x,
                _Float16* __restrict__ h) {
    __shared__ float red[256];
    const int row = blockIdx.x, tid = threadIdx.x;
    const _Float16* orr = oraw + (size_t)row * 512;
    const float*    xr  = x    + (size_t)row * 512;
    float v0 = (float)orr[tid]       + xr[tid];
    float v1 = (float)orr[tid + 256] + xr[tid + 256];

    red[tid] = v0 * v0 + v1 * v1;
    __syncthreads();
    for (int st = 128; st > 0; st >>= 1) { if (tid < st) red[tid] += red[tid + st]; __syncthreads(); }
    const float inv = 1.0f / fmaxf(sqrtf(red[0]), 1e-12f);

    h[(size_t)row * 512 + tid]       = (_Float16)(v0 * inv);
    h[(size_t)row * 512 + tid + 256] = (_Float16)(v1 * inv);
}

extern "C" void kernel_launch(void* const* d_in, const int* in_sizes, int n_in,
                              void* d_out, int out_size, void* d_ws, size_t ws_size,
                              hipStream_t stream) {
    const float* x      = (const float*)d_in[0];   // (16,2048,512)
    const float* source = (const float*)d_in[1];   // (16,2048,256)
    const float* Wq = (const float*)d_in[2];
    const float* bq = (const float*)d_in[3];
    const float* Wk = (const float*)d_in[4];
    const float* bk = (const float*)d_in[5];
    // d_in[6..8] = w_g, Wp, bp: dead code in the reference.
    const float* Wf = (const float*)d_in[9];
    const float* bf = (const float*)d_in[10];
    const float* W1 = (const float*)d_in[11];
    const float* b1 = (const float*)d_in[12];
    const float* W2 = (const float*)d_in[13];
    const float* b2 = (const float*)d_in[14];

    const int    Mr  = 16 * 2048;               // 32768 rows
    _Float16*    wsh = (_Float16*)d_ws;
    size_t o = 0;
    _Float16* xh  = wsh + o; o += (size_t)Mr * 512;
    _Float16* srh = wsh + o; o += (size_t)Mr * 256;
    _Float16* Wqt = wsh + o; o += 512 * 512;
    _Float16* Wkt = wsh + o; o += 512 * 256;    // [N=512][K=256]
    _Float16* Wft = wsh + o; o += 512 * 512;
    _Float16* W1t = wsh + o; o += 1024 * 512;   // [N=1024][K=512]
    _Float16* W2t = wsh + o; o += 512 * 1024;   // [N=512][K=1024]
    _Float16* qh  = wsh + o; o += (size_t)Mr * 512;   // later reused as oraw
    _Float16* kh  = wsh + o; o += (size_t)Mr * 512;   // later reused (in place) as s
    _Float16* hh  = wsh + o; o += (size_t)Mr * 512;
    _Float16* th  = wsh + o; o += (size_t)Mr * 1024;

    dim3 blk(256);

    // --- one-time precision/layout conversions (tiny vs 23.3 TB/s) ---
    cvt_f2h<<<(Mr * 512) / 1024, blk, 0, stream>>>(x, xh);
    cvt_f2h<<<(Mr * 256) / 1024, blk, 0, stream>>>(source, srh);
    transpose_f2h<<<(512 * 512) / 256, blk, 0, stream>>>(Wq, Wqt, 512, 512);
    transpose_f2h<<<(512 * 256) / 256, blk, 0, stream>>>(Wk, Wkt, 256, 512);
    transpose_f2h<<<(512 * 512) / 256, blk, 0, stream>>>(Wf, Wft, 512, 512);
    transpose_f2h<<<(1024 * 512) / 256, blk, 0, stream>>>(W1, W1t, 512, 1024);
    transpose_f2h<<<(512 * 1024) / 256, blk, 0, stream>>>(W2, W2t, 1024, 512);

    // --- pipeline ---
    gemm_wmma<EPI_NONE, _Float16><<<dim3(512 / BN, Mr / BM), blk, 0, stream>>>(
        xh, Wqt, bq, qh, Mr, 512, 512, nullptr);
    gemm_wmma<EPI_NONE, _Float16><<<dim3(512 / BN, Mr / BM), blk, 0, stream>>>(
        srh, Wkt, bk, kh, Mr, 512, 256, nullptr);
    l2norm_sum2<<<Mr, blk, 0, stream>>>(qh, kh, kh);          // s in place (row-local)
    gemm_wmma<EPI_NONE, _Float16><<<dim3(512 / BN, Mr / BM), blk, 0, stream>>>(
        kh, Wft, bf, qh, Mr, 512, 512, nullptr);              // oraw reuses qh
    add_l2norm<<<Mr, blk, 0, stream>>>(qh, x, hh);
    gemm_wmma<EPI_GELU, _Float16><<<dim3(1024 / BN, Mr / BM), blk, 0, stream>>>(
        hh, W1t, b1, th, Mr, 1024, 512, nullptr);
    gemm_wmma<EPI_ADD, float><<<dim3(512 / BN, Mr / BM), blk, 0, stream>>>(
        th, W2t, b2, (float*)d_out, Mr, 512, 1024, hh);
}
